// ReLeNet_23742579212507
// MI455X (gfx1250) — compile-verified, matrix operand order checked by reading, not run on hardware
//
#include <hip/hip_runtime.h>

// ---------------------------------------------------------------------------
// Types for CDNA5 WMMA
// ---------------------------------------------------------------------------
typedef __bf16 bf16;
typedef __attribute__((ext_vector_type(16))) __bf16 v16bf;
typedef __attribute__((ext_vector_type(8)))  float  v8f;

#define BM 128
#define BN 128
#define BK 32
#define LDSS 48   // LDS row stride in elements (96 bytes -> 32B aligned rows)

// ---------------------------------------------------------------------------
// Tensor Data Mover path (gfx1250 only, builtin-gated; fallback = manual stage)
// ---------------------------------------------------------------------------
#if defined(__gfx1250__) && __has_builtin(__builtin_amdgcn_tensor_load_to_lds)
#define GEMM_USE_TDM 1
#else
#define GEMM_USE_TDM 0
#endif

#if GEMM_USE_TDM
typedef unsigned int u32x4 __attribute__((ext_vector_type(4)));
typedef int          i32x4 __attribute__((ext_vector_type(4)));
typedef int          i32x8 __attribute__((ext_vector_type(8)));

// DMA a 128-row x 32-element bf16 tile (row stride K elements in memory) into
// LDS at byte offset lds_byte, padding each 64B row with 32B so the LDS row
// stride is LDSS (48) elements.  D# layout per CDNA5 ISA section 8.3/8.4.
__device__ __forceinline__ void tdm_load_tile(const bf16* gsrc, unsigned int lds_byte, int K)
{
    unsigned long long ga = (unsigned long long)(uintptr_t)gsrc;

    u32x4 g0;
    g0[0] = 1u;                                     // count=1 (valid user D#)
    g0[1] = lds_byte;                               // lds_addr
    g0[2] = (unsigned int)ga;                       // global_addr[31:0]
    g0[3] = (unsigned int)((ga >> 32) & 0x01ffffffu) | (2u << 30); // addr[56:32], type=2

    // data_size=2B (1<<16), pad_enable (1<<20), pad_interval=3 (16 DW = 64B row),
    // pad_amount=7 (8 DW = 32B pad) -> 96B LDS row stride = LDSS elements.
    i32x8 g1;
    unsigned int td0 = 0x40000000u, td1 = 0x40000000u;   // huge tensor dims: no OOB clamp
    unsigned long long s0 = (unsigned long long)(unsigned int)K; // row stride (elements)
    g1[0] = (int)((1u << 16) | (1u << 20) | (3u << 22) | (7u << 25));
    g1[1] = (int)((td0 & 0xffffu) << 16);                         // tensor_dim0[15:0]
    g1[2] = (int)(((td0 >> 16) & 0xffffu) | ((td1 & 0xffffu) << 16));
    g1[3] = (int)(((td1 >> 16) & 0xffffu) | (32u << 16));         // tile_dim0 = 32
    g1[4] = 128;                                                  // tile_dim1 = 128
    g1[5] = (int)(unsigned int)s0;                                // dim0_stride[31:0]
    g1[6] = (int)(unsigned int)(s0 >> 32);                        // dim0_stride[47:32]
    g1[7] = 0;

    i32x4 z4 = {0, 0, 0, 0};
#if __clang_major__ >= 23
    i32x8 z8 = {0, 0, 0, 0, 0, 0, 0, 0};
    __builtin_amdgcn_tensor_load_to_lds(g0, g1, z4, z4, z8, 0);
#else
    __builtin_amdgcn_tensor_load_to_lds(g0, g1, z4, z4, 0);
#endif
}
#endif // GEMM_USE_TDM

// ---------------------------------------------------------------------------
// Generic bf16 WMMA GEMM: C[M,N] = act( A[M,K] * B[N,K]^T + addend + bias )
// A, B bf16 row-major. C fp32 (optional), Cb bf16 (optional).
// Requires K % 32 == 0. M, N guarded on store.
// Double-buffered in LDS: TDM DMA for tile k+1 overlaps WMMA on tile k;
// s_wait_tensorcnt(2) lets the two in-flight prefetch DMAs ride while the
// current tile's DMAs (in-order per wave) are guaranteed complete.
// ---------------------------------------------------------------------------
__global__ __launch_bounds__(256)
void gemm_bf16_wmma(const bf16* __restrict__ A, const bf16* __restrict__ B,
                    const float* __restrict__ bias, const float* __restrict__ addend,
                    float* __restrict__ C, bf16* __restrict__ Cb,
                    int M, int N, int K, int act)
{
    __shared__ bf16 As[2][BM * LDSS];
    __shared__ bf16 Bs[2][BN * LDSS];

    const int tid   = threadIdx.x;
    const int lane  = tid & 31;
    const int wave  = tid >> 5;
    const int waveM = wave & 1;    // 2 waves along M  (64 rows each)
    const int waveN = wave >> 1;   // 4 waves along N  (32 cols each)

    const long blockRow = (long)blockIdx.y * BM;
    const long blockCol = (long)blockIdx.x * BN;

    v8f acc[4][2];
#pragma unroll
    for (int i = 0; i < 4; ++i)
#pragma unroll
        for (int j = 0; j < 2; ++j) acc[i][j] = (v8f){};

    const int loadRow = tid >> 1;        // 0..127
    const int loadSeg = (tid & 1) * 16;  // 0 or 16 elements
    (void)loadRow; (void)loadSeg;

    const int frow = lane & 15;
    const int fk   = (lane >> 4) * 16;

    const int nk = K >> 5;   // K / BK

#if GEMM_USE_TDM
    // Prologue: DMA tile 0 into buffer 0 (wave 0 only; D# is wave-uniform).
    if (tid < 32) {
        tdm_load_tile(A + blockRow * (long)K, (unsigned int)(uintptr_t)(void*)&As[0][0], K);
        tdm_load_tile(B + blockCol * (long)K, (unsigned int)(uintptr_t)(void*)&Bs[0][0], K);
    }
#endif

    for (int i = 0; i < nk; ++i) {
        const int cur = i & 1;
        const bf16* Acur = &As[cur][0];
        const bf16* Bcur = &Bs[cur][0];

#if GEMM_USE_TDM
        if (tid < 32) {
            if (i + 1 < nk) {
                const int nxt = cur ^ 1;
                const int k1  = (i + 1) * BK;
                tdm_load_tile(A + blockRow * (long)K + k1,
                              (unsigned int)(uintptr_t)(void*)&As[nxt][0], K);
                tdm_load_tile(B + blockCol * (long)K + k1,
                              (unsigned int)(uintptr_t)(void*)&Bs[nxt][0], K);
                // 2 prefetch DMAs may remain in flight; current tile's 2 are done.
                __builtin_amdgcn_s_wait_tensorcnt(2);
            } else {
                __builtin_amdgcn_s_wait_tensorcnt(0);
            }
        }
        __syncthreads();
#else
        // Manual staging fallback: 256 threads x 32B each per tile.
        {
            const int k0 = i * BK;
            const bf16* srcA = A + (blockRow + loadRow) * (long)K + k0 + loadSeg;
            const bf16* srcB = B + (blockCol + loadRow) * (long)K + k0 + loadSeg;
            float4 a0 = *(const float4*)(srcA);
            float4 a1 = *(const float4*)(srcA + 8);
            float4 b0 = *(const float4*)(srcB);
            float4 b1 = *(const float4*)(srcB + 8);
            *(float4*)&As[cur][loadRow * LDSS + loadSeg]     = a0;
            *(float4*)&As[cur][loadRow * LDSS + loadSeg + 8] = a1;
            *(float4*)&Bs[cur][loadRow * LDSS + loadSeg]     = b0;
            *(float4*)&Bs[cur][loadRow * LDSS + loadSeg + 8] = b1;
            if (k0 + BK < K) {  // speculative prefetch of next K tile
                __builtin_prefetch(srcA + BK, 0, 1);
                __builtin_prefetch(srcB + BK, 0, 1);
            }
        }
        __syncthreads();
#endif

        v16bf bfrag[2];
#pragma unroll
        for (int ni = 0; ni < 2; ++ni)
            bfrag[ni] = *(const v16bf*)&Bcur[(waveN * 32 + ni * 16 + frow) * LDSS + fk];

#pragma unroll
        for (int mi = 0; mi < 4; ++mi) {
            v16bf afrag = *(const v16bf*)&Acur[(waveM * 64 + mi * 16 + frow) * LDSS + fk];
#pragma unroll
            for (int ni = 0; ni < 2; ++ni) {
                acc[mi][ni] = __builtin_amdgcn_wmma_f32_16x16x32_bf16(
                    /*neg_a=*/false, afrag, /*neg_b=*/false, bfrag[ni],
                    /*c_mod=*/(short)0, acc[mi][ni],
                    /*reuse_a=*/false, /*reuse_b=*/false);
            }
        }
        // Protects buffer cur (the prefetch target of the *next* iteration)
        // from being overwritten before this iteration's LDS reads retire.
        __syncthreads();
    }

    // Epilogue. C/D layout: VGPR r -> M = r + 8*(lane/16); N = lane%16.
#pragma unroll
    for (int mi = 0; mi < 4; ++mi) {
        long rbase = blockRow + waveM * 64 + mi * 16 + (lane >> 4) * 8;
#pragma unroll
        for (int ni = 0; ni < 2; ++ni) {
            long col = blockCol + waveN * 32 + ni * 16 + (lane & 15);
#pragma unroll
            for (int r = 0; r < 8; ++r) {
                long row = rbase + r;
                if (row < M && col < N) {
                    long idx = row * (long)N + col;
                    float v = acc[mi][ni][r];
                    if (addend) v += addend[idx];
                    if (bias)   v += bias[col];
                    if (act)    v = v > 0.f ? v : 0.f;
                    if (C)  C[idx]  = v;
                    if (Cb) Cb[idx] = (bf16)v;
                }
            }
        }
    }
}

// ---------------------------------------------------------------------------
// Helper / glue kernels
// ---------------------------------------------------------------------------
__global__ void zero_f32(float* p, long n) {
    long i = blockIdx.x * (long)blockDim.x + threadIdx.x;
    if (i < n) p[i] = 0.f;
}

__global__ void vec_add(const float* a, const float* b, float* c, int n) {
    int i = blockIdx.x * blockDim.x + threadIdx.x;
    if (i < n) c[i] = a[i] + b[i];
}

__global__ void pad_bias(const float* src, float* dst, int n, int np) {
    int i = blockIdx.x * blockDim.x + threadIdx.x;
    if (i < np) dst[i] = (i < n) ? src[i] : 0.f;
}

// fp32 [N,K] -> bf16 [Np,Kp] with zero padding
__global__ void pack_w(const float* src, bf16* dst, int N, int K, int Np, int Kp) {
    long i = blockIdx.x * (long)blockDim.x + threadIdx.x;
    long tot = (long)Np * Kp;
    if (i >= tot) return;
    long n = i / Kp, k = i - n * Kp;
    float v = (n < N && k < K) ? src[n * (long)K + k] : 0.f;
    dst[i] = (bf16)v;
}

// X[512,3,32,32] -> Vfw bf16 [64, B, 64] (t = w*8+h, k = c*16+i*4+j, pad 48->64)
__global__ void pack_patches(const float* X, bf16* V, int B) {
    long i = blockIdx.x * (long)blockDim.x + threadIdx.x;
    long tot = 64L * B * 64;
    if (i >= tot) return;
    int kp = (int)(i & 63); long r = i >> 6;
    int b = (int)(r % B); int t = (int)(r / B);
    float v = 0.f;
    if (kp < 48) {
        int c = kp / 16, rem = kp % 16, ii = rem / 4, jj = rem % 4;
        int w = t / 8, h = t % 8;
        v = X[((long)b * 3 + c) * 1024 + (4 * h + ii) * 32 + (4 * w + jj)];
    }
    V[i] = (bf16)v;
}

// LSTM pointwise: gates fp32 [B,1280], c fp32 in/out, h bf16 out + sequence store
__global__ void lstm_step(const float* __restrict__ gates, float* __restrict__ c,
                          bf16* __restrict__ h, bf16* __restrict__ outp,
                          long out_stride, int B) {
    int i = blockIdx.x * blockDim.x + threadIdx.x;
    if (i >= B * 320) return;
    int b = i / 320, j = i - b * 320;
    const float* g = gates + (long)b * 1280;
    float ig = 1.f / (1.f + __expf(-g[j]));
    float fg = 1.f / (1.f + __expf(-g[320 + j]));
    float gg = tanhf(g[640 + j]);
    float og = 1.f / (1.f + __expf(-g[960 + j]));
    float cc = fg * c[i] + ig * gg;
    c[i] = cc;
    float hh = og * tanhf(cc);
    h[(long)b * 320 + j] = (bf16)hh;
    outp[(long)b * out_stride + j] = (bf16)hh;
}

// im2col for conv1: X -> col bf16 [(b,oh,ow), 32] (k = c*9+kh*3+kw, pad 27->32)
__global__ void im2col1(const float* X, bf16* col, int B) {
    long i = blockIdx.x * (long)blockDim.x + threadIdx.x;
    long tot = (long)B * 900 * 32;
    if (i >= tot) return;
    int k = (int)(i & 31); long r = i >> 5;
    int ow = (int)(r % 30); r /= 30;
    int oh = (int)(r % 30); int b = (int)(r / 30);
    float v = 0.f;
    if (k < 27) {
        int c = k / 9, kh = (k % 9) / 3, kw = k % 3;
        v = X[((long)b * 3 + c) * 1024 + (oh + kh) * 32 + (ow + kw)];
    }
    col[i] = (bf16)v;
}

// 2x2 maxpool on bf16 NHWC [B,30,30,128] -> fp32 [B,15,15,128]
__global__ void maxpool_bf16(const bf16* y, float* out, int B) {
    long i = blockIdx.x * (long)blockDim.x + threadIdx.x;
    long tot = (long)B * 225 * 128;
    if (i >= tot) return;
    int c = (int)(i & 127); long r = i >> 7;
    int pw = (int)(r % 15); r /= 15;
    int ph = (int)(r % 15); int b = (int)(r / 15);
    long base = ((long)b * 30) * 30 * 128;
    float a0 = (float)y[base + ((2 * ph) * 30 + 2 * pw) * 128 + c];
    float a1 = (float)y[base + ((2 * ph) * 30 + 2 * pw + 1) * 128 + c];
    float a2 = (float)y[base + ((2 * ph + 1) * 30 + 2 * pw) * 128 + c];
    float a3 = (float)y[base + ((2 * ph + 1) * 30 + 2 * pw + 1) * 128 + c];
    out[i] = fmaxf(fmaxf(a0, a1), fmaxf(a2, a3));
}

// Per-channel mean / rstd over `rows` samples with channel stride `stride`
__global__ __launch_bounds__(256)
void bn_stats(const float* x, float* mean, float* rstd, long rows, int stride) {
    int c = blockIdx.x, tid = threadIdx.x;
    float s = 0.f, s2 = 0.f;
    for (long r = tid; r < rows; r += 256) {
        float v = x[r * (long)stride + c];
        s += v; s2 += v * v;
    }
    __shared__ float sh[256], sh2[256];
    sh[tid] = s; sh2[tid] = s2; __syncthreads();
    for (int o = 128; o > 0; o >>= 1) {
        if (tid < o) { sh[tid] += sh[tid + o]; sh2[tid] += sh2[tid + o]; }
        __syncthreads();
    }
    if (tid == 0) {
        float m = sh[0] / (float)rows;
        float v = sh2[0] / (float)rows - m * m;
        mean[c] = m;
        rstd[c] = rsqrtf(v + 1e-5f);
    }
}

// Apply BN1 + im2col for conv2: pooled fp32 NHWC [B,15,15,128] ->
// col2 bf16 [(b,oh,ow), 2048] with k = c*16 + kh*4 + kw
__global__ void bn_im2col2(const float* pooled, const float* mean, const float* rstd,
                           const float* g, const float* bb, bf16* col, int B) {
    long i = blockIdx.x * (long)blockDim.x + threadIdx.x;
    long tot = (long)B * 144 * 2048;
    if (i >= tot) return;
    int k = (int)(i & 2047); long r = i >> 11;
    int ow = (int)(r % 12); r /= 12;
    int oh = (int)(r % 12); int b = (int)(r / 12);
    int c = k >> 4, kh = (k >> 2) & 3, kw = k & 3;
    float v = pooled[(((long)b * 15 + oh + kh) * 15 + (ow + kw)) * 128 + c];
    v = (v - mean[c]) * rstd[c] * g[c] + bb[c];
    col[i] = (bf16)v;
}

// Apply BN2 + repack NHWC [B,12,12,128(48 valid)] -> bf16 [B, 6912] (k=c*144+oh*12+ow)
__global__ void bn2_pack(const float* y2, const float* mean, const float* rstd,
                         const float* g, const float* bb, bf16* out, int B) {
    long i = blockIdx.x * (long)blockDim.x + threadIdx.x;
    long tot = (long)B * 6912;
    if (i >= tot) return;
    int k = (int)(i % 6912); int b = (int)(i / 6912);
    int c = k / 144, rem = k % 144, oh = rem / 12, ow = rem % 12;
    float v = y2[(((long)b * 12 + oh) * 12 + ow) * 128 + c];
    v = (v - mean[c]) * rstd[c] * g[c] + bb[c];
    out[i] = (bf16)v;
}

// fc (4096 -> 10) + log_softmax, one block per batch row
__global__ __launch_bounds__(256)
void fc_logsoftmax(const float* o1, const float* o2, const float* w, const float* b,
                   float* out) {
    int bidx = blockIdx.x, tid = threadIdx.x;
    float accv[10];
#pragma unroll
    for (int j = 0; j < 10; ++j) accv[j] = 0.f;
    const float* r1 = o1 + (long)bidx * 2048;
    const float* r2 = o2 + (long)bidx * 2048;
    for (int k = tid; k < 2048; k += 256) {
        float x1 = r1[k], x2 = r2[k];
#pragma unroll
        for (int j = 0; j < 10; ++j)
            accv[j] += x1 * w[j * 4096 + k] + x2 * w[j * 4096 + 2048 + k];
    }
    __shared__ float sh[256];
    __shared__ float logits[10];
    for (int j = 0; j < 10; ++j) {
        sh[tid] = accv[j]; __syncthreads();
        for (int o = 128; o > 0; o >>= 1) {
            if (tid < o) sh[tid] += sh[tid + o];
            __syncthreads();
        }
        if (tid == 0) logits[j] = sh[0] + b[j];
        __syncthreads();
    }
    if (tid == 0) {
        float mx = logits[0];
        for (int j = 1; j < 10; ++j) mx = fmaxf(mx, logits[j]);
        float se = 0.f;
        for (int j = 0; j < 10; ++j) se += __expf(logits[j] - mx);
        float lse = mx + __logf(se);
        for (int j = 0; j < 10; ++j) out[(long)bidx * 10 + j] = logits[j] - lse;
    }
}

// ---------------------------------------------------------------------------
// Host-side orchestration
// ---------------------------------------------------------------------------
static inline char* arena_take(char*& p, size_t bytes) {
    char* r = p;
    p += (bytes + 255) & ~(size_t)255;
    return r;
}

static void gemm(hipStream_t s, const bf16* A, const bf16* B, const float* bias,
                 const float* addend, float* C, bf16* Cb, long M, int N, int K, int act) {
    dim3 g((unsigned)((N + BN - 1) / BN), (unsigned)((M + BM - 1) / BM));
    gemm_bf16_wmma<<<g, 256, 0, s>>>(A, B, bias, addend, C, Cb, (int)M, N, K, act);
}

static inline dim3 grid1(long n) { return dim3((unsigned)((n + 255) / 256)); }

extern "C" void kernel_launch(void* const* d_in, const int* in_sizes, int n_in,
                              void* d_out, int out_size, void* d_ws, size_t ws_size,
                              hipStream_t stream) {
    (void)in_sizes; (void)n_in; (void)out_size; (void)ws_size;
    const int B = 512;

    const float* X       = (const float*)d_in[0];
    const float* conv1_w = (const float*)d_in[1];
    const float* conv1_b = (const float*)d_in[2];
    const float* bn1_g   = (const float*)d_in[3];
    const float* bn1_b   = (const float*)d_in[4];
    const float* conv2_w = (const float*)d_in[5];
    const float* conv2_b = (const float*)d_in[6];
    const float* bn2_g   = (const float*)d_in[7];
    const float* bn2_b   = (const float*)d_in[8];
    const float* rWih[4] = { (const float*)d_in[9],  (const float*)d_in[13],
                             (const float*)d_in[17], (const float*)d_in[21] };
    const float* rWhh[4] = { (const float*)d_in[10], (const float*)d_in[14],
                             (const float*)d_in[18], (const float*)d_in[22] };
    const float* rbih[4] = { (const float*)d_in[11], (const float*)d_in[15],
                             (const float*)d_in[19], (const float*)d_in[23] };
    const float* rbhh[4] = { (const float*)d_in[12], (const float*)d_in[16],
                             (const float*)d_in[20], (const float*)d_in[24] };
    const float* dense1_w = (const float*)d_in[25];
    const float* dense1_b = (const float*)d_in[26];
    const float* dense2_w = (const float*)d_in[27];
    const float* dense2_b = (const float*)d_in[28];
    const float* fc_w     = (const float*)d_in[29];
    const float* fc_b     = (const float*)d_in[30];
    float* out = (float*)d_out;

    // ---- workspace arena ----
    char* p = (char*)d_ws;
    bf16*  Vfw    = (bf16*) arena_take(p, 64L * B * 64 * 2);
    bf16*  Wihp[4]; bf16* Whhp[4]; float* biasr[4];
    Wihp[0] = (bf16*)arena_take(p, 1280L * 64  * 2);
    Wihp[1] = (bf16*)arena_take(p, 1280L * 64  * 2);
    Wihp[2] = (bf16*)arena_take(p, 1280L * 640 * 2);
    Wihp[3] = (bf16*)arena_take(p, 1280L * 640 * 2);
    for (int i = 0; i < 4; ++i) Whhp[i]  = (bf16*) arena_take(p, 1280L * 320 * 2);
    for (int i = 0; i < 4; ++i) biasr[i] = (float*)arena_take(p, 1280L * 4);
    bf16*  d1wp   = (bf16*) arena_take(p, 2048L * 40960 * 2);
    bf16*  d2wp   = (bf16*) arena_take(p, 2048L * 6912 * 2);
    bf16*  c1wp   = (bf16*) arena_take(p, 128L * 32 * 2);
    bf16*  c2wp   = (bf16*) arena_take(p, 128L * 2048 * 2);
    float* c2bpad = (float*)arena_take(p, 128L * 4);
    float* xg     = (float*)arena_take(p, 32768L * 1280 * 4);   // reused for all 4 LSTMs
    bf16*  H2     = (bf16*) arena_take(p, 64L * B * 640 * 2);
    bf16*  Hfm    = (bf16*) arena_take(p, (long)B * 40960 * 2);
    float* gates  = (float*)arena_take(p, (long)B * 1280 * 4);
    float* cst    = (float*)arena_take(p, (long)B * 320 * 4);
    bf16*  hst    = (bf16*) arena_take(p, (long)B * 320 * 2);
    bf16*  col1   = (bf16*) arena_take(p, 460800L * 32 * 2);
    bf16*  y1     = (bf16*) arena_take(p, 460800L * 128 * 2);
    float* pooled = (float*)arena_take(p, 115200L * 128 * 4);
    float* mean1  = (float*)arena_take(p, 128 * 4);
    float* rstd1  = (float*)arena_take(p, 128 * 4);
    bf16*  col2   = (bf16*) arena_take(p, 73728L * 2048 * 2);
    float* y2     = (float*)arena_take(p, 73728L * 128 * 4);
    float* mean2  = (float*)arena_take(p, 128 * 4);
    float* rstd2  = (float*)arena_take(p, 128 * 4);
    bf16*  d2in   = (bf16*) arena_take(p, (long)B * 6912 * 2);
    float* out1   = (float*)arena_take(p, (long)B * 2048 * 4);
    float* out2   = (float*)arena_take(p, (long)B * 2048 * 4);

    // ---- pack weights to bf16 (with padding) ----
    pack_w<<<grid1(1280L * 64),  256, 0, stream>>>(rWih[0], Wihp[0], 1280, 48,  1280, 64);
    pack_w<<<grid1(1280L * 64),  256, 0, stream>>>(rWih[1], Wihp[1], 1280, 48,  1280, 64);
    pack_w<<<grid1(1280L * 640), 256, 0, stream>>>(rWih[2], Wihp[2], 1280, 640, 1280, 640);
    pack_w<<<grid1(1280L * 640), 256, 0, stream>>>(rWih[3], Wihp[3], 1280, 640, 1280, 640);
    for (int i = 0; i < 4; ++i) {
        pack_w<<<grid1(1280L * 320), 256, 0, stream>>>(rWhh[i], Whhp[i], 1280, 320, 1280, 320);
        vec_add<<<grid1(1280), 256, 0, stream>>>(rbih[i], rbhh[i], biasr[i], 1280);
    }
    pack_w<<<grid1(2048L * 40960), 256, 0, stream>>>(dense1_w, d1wp, 2048, 40960, 2048, 40960);
    pack_w<<<grid1(2048L * 6912),  256, 0, stream>>>(dense2_w, d2wp, 2048, 6912,  2048, 6912);
    pack_w<<<grid1(128L * 32),     256, 0, stream>>>(conv1_w,  c1wp, 128, 27,   128, 32);
    pack_w<<<grid1(128L * 2048),   256, 0, stream>>>(conv2_w,  c2wp, 48,  2048, 128, 2048);
    pad_bias<<<1, 128, 0, stream>>>(conv2_b, c2bpad, 48, 128);

    // ---- patch extraction ----
    pack_patches<<<grid1(64L * B * 64), 256, 0, stream>>>(X, Vfw, B);

    // ---- LSTM 1 (forward over Vfw), writes H2[:, :, 0:320] ----
    gemm(stream, Vfw, Wihp[0], biasr[0], nullptr, xg, nullptr, 64L * B, 1280, 64, 0);
    zero_f32<<<grid1((long)B * 320), 256, 0, stream>>>(cst, (long)B * 320);
    zero_f32<<<grid1((long)B * 160), 256, 0, stream>>>((float*)hst, (long)B * 160);
    for (int t = 0; t < 64; ++t) {
        gemm(stream, hst, Whhp[0], nullptr, xg + (long)t * B * 1280, gates, nullptr, B, 1280, 320, 0);
        lstm_step<<<grid1((long)B * 320), 256, 0, stream>>>(gates, cst, hst,
            H2 + (long)t * B * 640, 640, B);
    }

    // ---- LSTM 2 (reversed sequence), writes H2[:, :, 320:640] ----
    gemm(stream, Vfw, Wihp[1], biasr[1], nullptr, xg, nullptr, 64L * B, 1280, 64, 0);
    zero_f32<<<grid1((long)B * 320), 256, 0, stream>>>(cst, (long)B * 320);
    zero_f32<<<grid1((long)B * 160), 256, 0, stream>>>((float*)hst, (long)B * 160);
    for (int t = 0; t < 64; ++t) {
        gemm(stream, hst, Whhp[1], nullptr, xg + (long)(63 - t) * B * 1280, gates, nullptr, B, 1280, 320, 0);
        lstm_step<<<grid1((long)B * 320), 256, 0, stream>>>(gates, cst, hst,
            H2 + (long)t * B * 640 + 320, 640, B);
    }

    // ---- LSTM 3 (forward over H2), writes Hfm[b, t*640 + 0:320] ----
    gemm(stream, H2, Wihp[2], biasr[2], nullptr, xg, nullptr, 64L * B, 1280, 640, 0);
    zero_f32<<<grid1((long)B * 320), 256, 0, stream>>>(cst, (long)B * 320);
    zero_f32<<<grid1((long)B * 160), 256, 0, stream>>>((float*)hst, (long)B * 160);
    for (int t = 0; t < 64; ++t) {
        gemm(stream, hst, Whhp[2], nullptr, xg + (long)t * B * 1280, gates, nullptr, B, 1280, 320, 0);
        lstm_step<<<grid1((long)B * 320), 256, 0, stream>>>(gates, cst, hst,
            Hfm + (long)t * 640, 40960, B);
    }

    // ---- LSTM 4 (reversed over H2), writes Hfm[b, t*640 + 320:640] ----
    gemm(stream, H2, Wihp[3], biasr[3], nullptr, xg, nullptr, 64L * B, 1280, 640, 0);
    zero_f32<<<grid1((long)B * 320), 256, 0, stream>>>(cst, (long)B * 320);
    zero_f32<<<grid1((long)B * 160), 256, 0, stream>>>((float*)hst, (long)B * 160);
    for (int t = 0; t < 64; ++t) {
        gemm(stream, hst, Whhp[3], nullptr, xg + (long)(63 - t) * B * 1280, gates, nullptr, B, 1280, 320, 0);
        lstm_step<<<grid1((long)B * 320), 256, 0, stream>>>(gates, cst, hst,
            Hfm + (long)t * 640 + 320, 40960, B);
    }

    // ---- dense1: [512, 40960] x [2048, 40960]^T + bias, ReLU ----
    gemm(stream, Hfm, d1wp, dense1_b, nullptr, out1, nullptr, B, 2048, 40960, 1);

    // ---- conv branch ----
    im2col1<<<grid1((long)B * 900 * 32), 256, 0, stream>>>(X, col1, B);
    gemm(stream, col1, c1wp, conv1_b, nullptr, nullptr, y1, (long)B * 900, 128, 32, 1);
    maxpool_bf16<<<grid1((long)B * 225 * 128), 256, 0, stream>>>(y1, pooled, B);
    bn_stats<<<128, 256, 0, stream>>>(pooled, mean1, rstd1, (long)B * 225, 128);
    bn_im2col2<<<grid1((long)B * 144 * 2048), 256, 0, stream>>>(pooled, mean1, rstd1,
                                                                bn1_g, bn1_b, col2, B);
    gemm(stream, col2, c2wp, c2bpad, nullptr, y2, nullptr, (long)B * 144, 128, 2048, 1);
    bn_stats<<<48, 256, 0, stream>>>(y2, mean2, rstd2, (long)B * 144, 128);
    bn2_pack<<<grid1((long)B * 6912), 256, 0, stream>>>(y2, mean2, rstd2, bn2_g, bn2_b, d2in, B);
    gemm(stream, d2in, d2wp, dense2_b, nullptr, out2, nullptr, B, 2048, 6912, 1);

    // ---- final FC + log_softmax ----
    fc_logsoftmax<<<B, 256, 0, stream>>>(out1, out2, fc_w, fc_b, out);
}
